// LocalTemporalMGMQEncoder_5497558139089
// MI455X (gfx1250) — matrix operand order repaired and checked by reading, not compiled, not run on hardware
//
#include <hip/hip_runtime.h>

typedef __attribute__((ext_vector_type(16))) _Float16 v16h;
typedef __attribute__((ext_vector_type(8)))  float    v8f;

#define B_     4096
#define K_     4
#define T_     5
#define OBS_   48
#define GSELF  (B_*T_)           /* 20480 self graphs   */
#define GNBR   (B_*K_*T_)        /* 81920 nbr graphs    */
#define GTOT   (GSELF+GNBR)      /* 102400 graphs       */
#define ROWS   (B_*T_*(1+K_))    /* 102400 sage rows    */

// ---------------------------------------------------------------------------
// WMMA fragment loaders (f16 16x16x32, wave32), contiguous k order (k = 16*(lane>>4) + s) in A and B.
// A (16xK=32): lane m = lane&15; half-wave selects K sub-block of 16;
//   elements 0..15 -> K = kb..kb+15 (kb = (lane>>4)*16).
//   Two contiguous 16B runs -> 2x ds_load_b128.
// B: staged PRE-SWIZZLED in LDS: tile t, lane L -> halves at (t*32+L)*16,
//   holding K = (L>>4)*16 + i, N = L&15. One aligned 32B v16h load.
// D/C v8f: row = j + 8*(lane>>4), col = lane&15.
// ---------------------------------------------------------------------------
__device__ inline v16h ldsA_frag(const _Float16* base, int lda, int m0, int k0, int lane) {
  int m  = m0 + (lane & 15);
  int kb = (lane >> 4) * 16;
  v16h a;
#pragma unroll
  for (int i = 0; i < 8; ++i) {
    a[i]     = base[m * lda + k0 + kb + i];
    a[i + 8] = base[m * lda + k0 + kb + 8 + i];
  }
  return a;
}

__device__ inline v16h ldsBsw_frag(const _Float16* base, int tile, int lane) {
  return *(const v16h*)(base + ((size_t)(tile * 32 + lane)) * 16);
}

__device__ inline v8f vzero8() {
  v8f z;
#pragma unroll
  for (int i = 0; i < 8; ++i) z[i] = 0.f;
  return z;
}

// ---------------------------------------------------------------------------
// Kernel 1: fused GAT-temporal encoder. One 256-thread WG handles 4 graphs
// (48 rows = 3 exact 16-row WMMA tiles). Emits emb[GTOT][256].
// ---------------------------------------------------------------------------
__global__ __launch_bounds__(256) void gat_kernel(
    const float* __restrict__ selfF, const float* __restrict__ nbrF,
    const float* __restrict__ Win,   const float* __restrict__ bin,
    const float* __restrict__ lng,   const float* __restrict__ lnb,
    const float* __restrict__ Wcoop, const float* __restrict__ aSc,
    const float* __restrict__ aDc,   const float* __restrict__ Wconf,
    const float* __restrict__ aSf,   const float* __restrict__ aDf,
    const float* __restrict__ adjc,  const float* __restrict__ adjf,
    float* __restrict__ emb)
{
  __shared__ float    sX[4 * 48];        // raw lane features, 4 graphs
  __shared__ _Float16 sHA[48 * 64];      // layernormed h (f16, GEMM A)
  __shared__ _Float16 sWc[16 * 32 * 16]; // K-half of Wcat, fragment-swizzled
  __shared__ _Float16 sWh[48 * 256];     // Wh result (f16); fp32 h aliased here first
  __shared__ float    sAS[8 * 32];       // a_src per head
  __shared__ float    sAD[8 * 32];       // a_dst per head
  __shared__ float    sAdj[2 * 144];     // coop / conf adjacency
  __shared__ float    sS[48], sD[48];    // attention logits src/dst
  __shared__ float    sAttn[4 * 144];    // softmax rows, 4 graphs
  __shared__ float    sEmb[4 * 256];     // pooled output

  const int tid  = threadIdx.x;
  const int lane = tid & 31;
  const int wave = tid >> 5;
  const int g0   = blockIdx.x * 4;

  // warm L2 with the shared weight tensors (global_prefetch_b8)
  __builtin_prefetch(&Wcoop[tid * 32], 0, 2);
  __builtin_prefetch(&Wconf[tid * 32], 0, 2);

  // ---- stage inputs -------------------------------------------------------
  for (int i = tid; i < 4 * 48; i += 256) {
    int gi = i / 48, e = i % 48, g = g0 + gi;
    sX[i] = (g < GSELF) ? selfF[(size_t)g * OBS_ + e]
                        : nbrF[(size_t)(g - GSELF) * OBS_ + e];
  }
  for (int i = tid; i < 8 * 32; i += 256) {
    int h = i >> 5, o = i & 31;
    sAS[i] = (h < 4) ? aSc[h * 32 + o] : aSf[(h - 4) * 32 + o];
    sAD[i] = (h < 4) ? aDc[h * 32 + o] : aDf[(h - 4) * 32 + o];
  }
  for (int i = tid; i < 288; i += 256)
    sAdj[i] = (i < 144) ? adjc[i] : adjf[i - 144];
  __syncthreads();

  // ---- h = x @ W_in + b_in  (K=4, VALU; fp32 result aliased into sWh) -----
  float* sH32 = reinterpret_cast<float*>(sWh);   // 48*64 floats (12KB of 24KB)
  for (int i = tid; i < 48 * 64; i += 256) {
    int row = i >> 6, col = i & 63;
    int gi = row / 12, ln = row % 12;
    const float* xr = &sX[gi * 48 + ln * 4];
    float acc = bin[col];
#pragma unroll
    for (int k = 0; k < 4; ++k) acc += xr[k] * Win[k * 64 + col];
    sH32[i] = acc;
  }
  __syncthreads();

  // ---- layernorm + relu -> f16 A matrix -----------------------------------
  if (tid < 48) {
    const float* r = &sH32[tid * 64];
    float mu = 0.f;
    for (int c = 0; c < 64; ++c) mu += r[c];
    mu *= (1.f / 64.f);
    float var = 0.f;
    for (int c = 0; c < 64; ++c) { float d = r[c] - mu; var += d * d; }
    var *= (1.f / 64.f);
    float rs = rsqrtf(var + 1e-5f);
    for (int c = 0; c < 64; ++c) {
      float v = (r[c] - mu) * rs * lng[c] + lnb[c];
      sHA[tid * 64 + c] = (_Float16)(v > 0.f ? v : 0.f);
    }
  }
  __syncthreads();

  // ---- WMMA GEMM: Wh[48,256] = hA[48,64] @ Wcat[64,256] -------------------
  // 48 tile-tasks (3 M-tiles x 16 N-tiles), 8 waves -> 6 tasks each; K in 2 halves.
  v8f acc[6];
#pragma unroll
  for (int j = 0; j < 6; ++j) acc[j] = vzero8();

  for (int kh = 0; kh < 2; ++kh) {
    __syncthreads();
    // stage B K-half (f16), written directly in fragment-swizzled order:
    // dst idx = ((nt*32 + ln)*16 + ii);  kk = (ln>>4)*16+ii, n = ln&15
    for (int i = tid; i < 16 * 32 * 16; i += 256) {
      int nt  = i >> 9;
      int ln  = (i >> 4) & 31;
      int ii  = i & 15;
      int kk  = ((ln >> 4) << 4) + ii;
      int c   = nt * 16 + (ln & 15);
      int gk  = kh * 32 + kk;
      float w = (c < 128) ? Wcoop[(c >> 5) * 2048 + gk * 32 + (c & 31)]
                          : Wconf[((c - 128) >> 5) * 2048 + gk * 32 + (c & 31)];
      sWc[i] = (_Float16)w;
    }
    __syncthreads();
#pragma unroll
    for (int j = 0; j < 6; ++j) {
      int tsk = wave + 8 * j;
      int mt = tsk >> 4, nt = tsk & 15;
      v16h a = ldsA_frag(sHA, 64, mt * 16, kh * 32, lane);
      v16h b = ldsBsw_frag(sWc, nt, lane);
      acc[j] = __builtin_amdgcn_wmma_f32_16x16x32_f16(
                   false, a, false, b, (short)0, acc[j], false, false);
    }
  }
  __syncthreads();   // h32 alias dead; now safe to overwrite with f16 Wh

#pragma unroll
  for (int j = 0; j < 6; ++j) {
    int tsk = wave + 8 * j;
    int mt = tsk >> 4, nt = tsk & 15;
    int col   = nt * 16 + (lane & 15);
    int rbase = mt * 16 + 8 * (lane >> 4);
#pragma unroll
    for (int jj = 0; jj < 8; ++jj)
      sWh[(rbase + jj) * 256 + col] = (_Float16)acc[j][jj];
  }
  __syncthreads();

  // ---- per-head attention: leaky-relu, mask, softmax, attn@Wh, ELU --------
  for (int h = 0; h < 8; ++h) {
    const int hcol = h * 32;
    const float* adjp = &sAdj[(h >> 2) * 144];

    if (tid < 96) {                       // 48 (graph,lane) x {src,dst} dots
      int which = tid / 48, r = tid % 48;
      const float* av = which ? &sAD[h * 32] : &sAS[h * 32];
      float s = 0.f;
      for (int o = 0; o < 32; ++o) s += (float)sWh[r * 256 + hcol + o] * av[o];
      if (which) sD[r] = s; else sS[r] = s;
    }
    __syncthreads();

    if (tid < 48) {                       // softmax over m for each (gi,n)
      int gi = tid / 12, n = tid % 12;
      float e[12], mx = -3.4e38f;
      for (int m = 0; m < 12; ++m) {
        float x = sS[gi * 12 + n] + sD[gi * 12 + m];
        x = (x > 0.f) ? x : 0.2f * x;                 // leaky_relu 0.2
        x = (adjp[n * 12 + m] > 0.f) ? x : -1e9f;     // mask
        e[m] = x; mx = fmaxf(mx, x);
      }
      float den = 0.f;
      for (int m = 0; m < 12; ++m) { e[m] = __expf(e[m] - mx); den += e[m]; }
      float inv = 1.f / den;
      for (int m = 0; m < 12; ++m) sAttn[gi * 144 + n * 12 + m] = e[m] * inv;
    }
    __syncthreads();

    if (tid < 128) {                      // out = elu(attn@Wh), mean over lanes
      int gi = tid >> 5, o = tid & 31;
      float accv = 0.f;
      for (int n = 0; n < 12; ++n) {
        float t = 0.f;
        for (int m = 0; m < 12; ++m)
          t += sAttn[gi * 144 + n * 12 + m] * (float)sWh[(gi * 12 + m) * 256 + hcol + o];
        accv += (t > 0.f) ? t : (__expf(t) - 1.f);    // elu
      }
      sEmb[gi * 256 + hcol + o] = accv * (1.f / 12.f);
    }
    __syncthreads();
  }

  for (int i = tid; i < 4 * 256; i += 256) {
    int gi = i >> 8, c = i & 255;
    emb[(size_t)(g0 + gi) * 256 + c] = sEmb[i];
  }
}

// ---------------------------------------------------------------------------
// Kernel 2: GraphSAGE as one GEMM: [node|agg](16,512) @ [Wself;Wneigh](512,64)
// 128 threads / 4 waves per 16-row tile; weight staged in 128-K chunks,
// fragment-swizzled (4 k-steps x 4 n-tiles).
// ---------------------------------------------------------------------------
__global__ __launch_bounds__(128) void sage_kernel(
    const float* __restrict__ emb, const float* __restrict__ Wself,
    const float* __restrict__ Wneigh, const float* __restrict__ bsage,
    const int* __restrict__ nmask, float* __restrict__ hs)
{
  __shared__ _Float16 sA[16 * 512];
  __shared__ _Float16 sW[4 * 4 * 32 * 16];   // [ks][nt][lane][16]
  __shared__ float    sBias[64];

  const int tid  = threadIdx.x;
  const int lane = tid & 31;
  const int wave = tid >> 5;
  const int r0   = blockIdx.x * 16;

  __builtin_prefetch(&Wself[tid * 128], 0, 2);
  __builtin_prefetch(&Wneigh[tid * 128], 0, 2);

  if (tid < 64) sBias[tid] = bsage[tid];

  {  // build A rows: 16 rows x 8 column-chunks of 64
    int lr = tid >> 3, chunk = tid & 7;
    int r  = r0 + lr;
    int bt = r / 5, n = r % 5;
    int b  = bt / T_, t = bt % T_;
    float mk[4]; float degsum = 0.f;
#pragma unroll
    for (int j = 0; j < 4; ++j) { mk[j] = (float)nmask[b * 4 + j]; degsum += mk[j]; }
    float invdeg0 = 1.f / fmaxf(degsum, 1.f);
    size_t selfRow = ((size_t)b * T_ + t) * 256;
    int c0 = chunk * 64;
    if (c0 < 256) {                       // node features
      size_t nodeRow = (n == 0) ? selfRow
          : ((size_t)GSELF + ((size_t)b * K_ + (n - 1)) * T_ + t) * 256;
      for (int i = 0; i < 64; ++i)
        sA[lr * 512 + c0 + i] = (_Float16)emb[nodeRow + c0 + i];
    } else {                              // mean-aggregated neighbor features
      int c = c0 - 256;
      if (n == 0) {
        for (int i = 0; i < 64; ++i) {
          float s = 0.f;
          for (int j = 0; j < 4; ++j) {
            if (mk[j] > 0.f) {
              size_t nr = ((size_t)GSELF + ((size_t)b * K_ + j) * T_ + t) * 256;
              s += emb[nr + c + i];
            }
          }
          sA[lr * 512 + c0 + i] = (_Float16)(s * invdeg0);
        }
      } else {
        float m = mk[n - 1];              // deg is 1 either way for leaves
        for (int i = 0; i < 64; ++i)
          sA[lr * 512 + c0 + i] = (_Float16)(m * emb[selfRow + c + i]);
      }
    }
  }

  v8f acc = vzero8();
  for (int kc = 0; kc < 4; ++kc) {
    __syncthreads();
    // stage 128-K weight chunk, fragment-swizzled:
    // dst idx = (((ks*4 + nt)*32 + ln)*16 + ii)
    for (int i = tid; i < 4 * 4 * 32 * 16; i += 128) {
      int ks = i >> 11;
      int nt = (i >> 9) & 3;
      int ln = (i >> 4) & 31;
      int ii = i & 15;
      int kk = ((ln >> 4) << 4) + ii;
      int gk = kc * 128 + ks * 32 + kk;
      int c  = nt * 16 + (ln & 15);
      float w = (gk < 256) ? Wself[gk * 64 + c] : Wneigh[(gk - 256) * 64 + c];
      sW[i] = (_Float16)w;
    }
    __syncthreads();
#pragma unroll
    for (int ks = 0; ks < 4; ++ks) {
      v16h a = ldsA_frag(sA, 512, 0, kc * 128 + ks * 32, lane);
      v16h b = ldsBsw_frag(sW, ks * 4 + wave, lane);
      acc = __builtin_amdgcn_wmma_f32_16x16x32_f16(
                false, a, false, b, (short)0, acc, false, false);
    }
  }

  int col   = wave * 16 + (lane & 15);
  int rbase = 8 * (lane >> 4);
#pragma unroll
  for (int jj = 0; jj < 8; ++jj) {
    float v = acc[jj] + sBias[col];
    hs[(size_t)(r0 + rbase + jj) * 64 + col] = v > 0.f ? v : 0.f;
  }
}

// ---------------------------------------------------------------------------
// Kernel 3: mean over the 5 star-graph nodes -> pooled[B*T][64]
// ---------------------------------------------------------------------------
__global__ void pool_kernel(const float* __restrict__ hs, float* __restrict__ pooled) {
  int idx = blockIdx.x * blockDim.x + threadIdx.x;
  if (idx >= GSELF * 64) return;
  int bt = idx >> 6, c = idx & 63;
  float s = 0.f;
#pragma unroll
  for (int n = 0; n < 5; ++n) s += hs[(size_t)(bt * 5 + n) * 64 + c];
  pooled[idx] = s * 0.2f;
}

// ---------------------------------------------------------------------------
// Kernel 4: bidirectional GRU (one wave per batch row, lane = hidden unit)
// plus final output assembly: [self_emb(t=T-1) | hf | hb]
// ---------------------------------------------------------------------------
__global__ __launch_bounds__(128) void gru_kernel(
    const float* __restrict__ pooled, const float* __restrict__ emb,
    const float* __restrict__ Wxf, const float* __restrict__ Whf,
    const float* __restrict__ bxf, const float* __restrict__ bhf,
    const float* __restrict__ Wxb, const float* __restrict__ Whb,
    const float* __restrict__ bxb, const float* __restrict__ bhb,
    float* __restrict__ out)
{
  __shared__ float sXb[4][5 * 64];
  __shared__ float sH[4][32];

  const int tid  = threadIdx.x;
  const int lane = tid & 31;
  const int wv   = tid >> 5;
  const int b    = blockIdx.x * 4 + wv;

  for (int i = lane; i < 5 * 64; i += 32)
    sXb[wv][i] = pooled[(size_t)b * T_ * 64 + i];

  float res_f = 0.f, res_b = 0.f;
  for (int dir = 0; dir < 2; ++dir) {
    const float* Wx = dir ? Wxb : Wxf;
    const float* Wh = dir ? Whb : Whf;
    const float* bx = dir ? bxb : bxf;
    const float* bh = dir ? bhb : bhf;
    float h = 0.f;
    sH[wv][lane] = 0.f;
    __syncthreads();
    for (int step = 0; step < 5; ++step) {
      int t = dir ? (4 - step) : step;
      const float* x = &sXb[wv][t * 64];
      float gr = bx[lane], gz = bx[32 + lane], gn = bx[64 + lane];
      for (int i = 0; i < 64; ++i) {
        float xi = x[i];
        gr += xi * Wx[i * 96 + lane];
        gz += xi * Wx[i * 96 + 32 + lane];
        gn += xi * Wx[i * 96 + 64 + lane];
      }
      float hr = bh[lane], hz = bh[32 + lane], hn = bh[64 + lane];
      for (int i = 0; i < 32; ++i) {
        float hi = sH[wv][i];
        hr += hi * Wh[i * 96 + lane];
        hz += hi * Wh[i * 96 + 32 + lane];
        hn += hi * Wh[i * 96 + 64 + lane];
      }
      float r = 1.f / (1.f + __expf(-(gr + hr)));
      float z = 1.f / (1.f + __expf(-(gz + hz)));
      float n = tanhf(gn + r * hn);
      h = (1.f - z) * n + z * h;
      __syncthreads();
      sH[wv][lane] = h;
      __syncthreads();
    }
    if (dir) res_b = h; else res_f = h;
  }

  out[(size_t)b * 320 + 256 + lane] = res_f;
  out[(size_t)b * 320 + 288 + lane] = res_b;
  size_t er = ((size_t)b * T_ + (T_ - 1)) * 256;   // self_emb[:, -1, :]
  for (int i = lane; i < 256; i += 32)
    out[(size_t)b * 320 + i] = emb[er + i];
}

// ---------------------------------------------------------------------------
extern "C" void kernel_launch(void* const* d_in, const int* in_sizes, int n_in,
                              void* d_out, int out_size, void* d_ws, size_t ws_size,
                              hipStream_t stream)
{
  const float* selfF = (const float*)d_in[0];
  const float* nbrF  = (const float*)d_in[1];
  const float* Win   = (const float*)d_in[2];
  const float* bin   = (const float*)d_in[3];
  const float* lng   = (const float*)d_in[4];
  const float* lnb   = (const float*)d_in[5];
  const float* Wcoop = (const float*)d_in[6];
  const float* aSc   = (const float*)d_in[7];
  const float* aDc   = (const float*)d_in[8];
  const float* Wconf = (const float*)d_in[9];
  const float* aSf   = (const float*)d_in[10];
  const float* aDf   = (const float*)d_in[11];
  const float* Wself = (const float*)d_in[12];
  const float* Wneigh= (const float*)d_in[13];
  const float* bsage = (const float*)d_in[14];
  const float* Wxf   = (const float*)d_in[15];
  const float* Whf   = (const float*)d_in[16];
  const float* bxf   = (const float*)d_in[17];
  const float* bhf   = (const float*)d_in[18];
  const float* Wxb   = (const float*)d_in[19];
  const float* Whb   = (const float*)d_in[20];
  const float* bxb   = (const float*)d_in[21];
  const float* bhb   = (const float*)d_in[22];
  const float* adjc  = (const float*)d_in[23];
  const float* adjf  = (const float*)d_in[24];
  const int*   nmask = (const int*)d_in[25];
  float* out = (float*)d_out;

  float* emb    = (float*)d_ws;                 // GTOT*256  (~104.9 MB)
  float* hs     = emb + (size_t)GTOT * 256;     // ROWS*64   (~26.2 MB)
  float* pooled = hs  + (size_t)ROWS * 64;      // B*T*64    (~5.2 MB)

  gat_kernel<<<GTOT / 4, 256, 0, stream>>>(selfF, nbrF, Win, bin, lng, lnb,
      Wcoop, aSc, aDc, Wconf, aSf, aDf, adjc, adjf, emb);
  sage_kernel<<<ROWS / 16, 128, 0, stream>>>(emb, Wself, Wneigh, bsage, nmask, hs);
  pool_kernel<<<(GSELF * 64 + 255) / 256, 256, 0, stream>>>(hs, pooled);
  gru_kernel<<<B_ / 4, 128, 0, stream>>>(pooled, emb, Wxf, Whf, bxf, bhf,
      Wxb, Whb, bxb, bhb, out);
}
